// GSC_DH_SNN_86474871538255
// MI455X (gfx1250) — compile-verified, matrix operand order checked by reading, not run on hardware
//
#include <hip/hip_runtime.h>
#include <hip/hip_bf16.h>

// ---------------------------------------------------------------------------
// GSC DH-SNN forward: 3 LIF layers + leaky readout, T=100 scan, log_softmax.
// B=1024, M=120 (pad->128), H=512, D_out=12 (pad->16), T=100.
// Strategy: 64 persistent blocks, each owns 16 batch rows for all T steps.
// Membrane states live in WMMA f32 accumulator VGPRs; spikes go through LDS
// as f16 A-matrices; weights pre-packed to WMMA B-fragment layout in d_ws.
// An opaque per-timestep SGPR offset (value 0) is added to the weight bases
// so loads are re-issued from L2 each step (no cross-t hoist/spill) while
// keeping global address space => global_load_b128, not flat_load.
// ---------------------------------------------------------------------------

typedef __attribute__((ext_vector_type(16))) _Float16 v16h;
typedef __attribute__((ext_vector_type(8)))  _Float16 v8h;
typedef __attribute__((ext_vector_type(8)))  float    v8f;

#define B_TOT   1024
#define TSTEPS  100
#define M_IN    120
#define KP1     128      // padded K for layer 1
#define HDIM    512
#define DOUT    12
#define LSTR    520      // LDS row stride (elements) for 16x512 spike tiles

// f16 WMMA wrapper (8-arg form, probe-verified)
__device__ __forceinline__ v8f wmma_f16(v16h a, v16h b, v8f c) {
  return __builtin_amdgcn_wmma_f32_16x16x32_f16(
      /*neg_a=*/false, a, /*neg_b=*/false, b,
      /*c_mod=*/(short)0, c, /*reuse_a=*/false, /*reuse_b=*/false);
}

// Opaque zero in an SGPR: blocks LICM of loads addressed with it, but keeps
// the base pointer's (global) address space intact.
__device__ __forceinline__ size_t opaque_zero() {
  size_t z = 0;
  asm volatile("" : "+s"(z));
  return z;
}

// Packed B-fragment index for element (k, n) of a KxN weight matrix.
// Matches V_WMMA_F32_16X16X32_F16 B layout (N on lanes 0-15 mirrored to
// 16-31, K split 0-7/16-23 vs 8-15/24-31 across the lane halves).
__host__ __device__ __forceinline__ int pack_idx(int k, int n, int Kp) {
  int nt   = n >> 4;
  int nl   = n & 15;
  int kk   = k >> 5;
  int kl   = k & 31;
  int grp  = kl >> 3;                 // 0..3
  int lane = nl + ((grp & 1) << 4);
  int i    = (kl & 7) + ((grp >> 1) << 3);
  return ((nt * (Kp >> 5) + kk) * 32 + lane) * 16 + i;
}

// ---------------------------------------------------------------------------
// Kernel 0: convert f32 weights -> f16, scatter into WMMA B-fragment layout.
// Workspace layout (f16):
//   W1p : KP1*HDIM  = 65536
//   W2p : HDIM*HDIM = 262144
//   W3p : HDIM*HDIM = 262144
//   Wrp : HDIM*16   = 8192       (total 598016 f16 = 1.17 MB, L2-resident)
// ---------------------------------------------------------------------------
__global__ void prep_pack(const float* __restrict__ W1,
                          const float* __restrict__ W2,
                          const float* __restrict__ W3,
                          const float* __restrict__ Wr,
                          _Float16* __restrict__ ws) {
  const int R1 = KP1 * HDIM;
  const int R2 = R1 + HDIM * HDIM;
  const int R3 = R2 + HDIM * HDIM;
  const int R4 = R3 + HDIM * 16;

  _Float16* W1p = ws;
  _Float16* W2p = ws + R1;
  _Float16* W3p = ws + R2;
  _Float16* Wrp = ws + R3;

  int tid = blockIdx.x * blockDim.x + threadIdx.x;
  if (tid < R1) {
    int k = tid >> 9, n = tid & 511;
    float v = (k < M_IN) ? W1[k * HDIM + n] : 0.0f;
    W1p[pack_idx(k, n, KP1)] = (_Float16)v;
  } else if (tid < R2) {
    int s = tid - R1;
    int k = s >> 9, n = s & 511;
    W2p[pack_idx(k, n, HDIM)] = (_Float16)W2[k * HDIM + n];
  } else if (tid < R3) {
    int s = tid - R2;
    int k = s >> 9, n = s & 511;
    W3p[pack_idx(k, n, HDIM)] = (_Float16)W3[k * HDIM + n];
  } else if (tid < R4) {
    int s = tid - R3;
    int k = s >> 4, n = s & 15;
    float v = (n < DOUT) ? Wr[k * DOUT + n] : 0.0f;
    Wrp[pack_idx(k, n, HDIM)] = (_Float16)v;
  }
}

// ---------------------------------------------------------------------------
// A-fragment loader from LDS (row-major 16 x K tile, row stride LSTR).
// 16-bit A 16x32 layout: lane L<16 holds row M=L, K {0..7,16..23};
// lane L>=16 holds row M=L-16, K {8..15,24..31}. Two 16B LDS loads.
// ---------------------------------------------------------------------------
__device__ __forceinline__ v16h load_a_frag(const _Float16* buf, int lane, int kk) {
  int row = lane & 15;
  int k0  = (kk << 5) + ((lane >> 4) << 3);
  const v8h* p0 = (const v8h*)(buf + row * LSTR + k0);
  const v8h* p1 = (const v8h*)(buf + row * LSTR + k0 + 16);
  v8h lo = *p0;
  v8h hi = *p1;
  return __builtin_shufflevector(lo, hi, 0, 1, 2, 3, 4, 5, 6, 7,
                                 8, 9, 10, 11, 12, 13, 14, 15);
}

// LIF on one 16x16 f32 C fragment, spike (0/1) stored to LDS as f16.
// C layout: VGPR r, lanes 0-15 -> (M=r, N=lane); lanes 16-31 -> (M=r+8, N=lane-16).
__device__ __forceinline__ void lif_store(v8f h, float bias, v8f& vst,
                                          _Float16* outBuf, int nt, int lane) {
  int colbase = nt * 16 + (lane & 15);
  int rbase   = (lane >> 4) << 3;
#pragma unroll
  for (int r = 0; r < 8; ++r) {
    float v = 0.5f * (vst[r] + h[r] + bias);   // v + (i - v)/tau, tau=2
    bool  fire = (v >= 1.0f);
    vst[r] = fire ? 0.0f : v;                  // hard reset
    outBuf[(rbase + r) * LSTR + colbase] = (_Float16)(fire ? 1.0f : 0.0f);
  }
}

// One hidden layer: C = A(16xK from LDS) @ Wpacked + b, LIF, spikes -> LDS.
// wP must carry a per-timestep opaque offset at the call site.
template <int KST>
__device__ __forceinline__ void layer_fwd(const _Float16* aBuf,
                                          const _Float16* __restrict__ wP,
                                          float bv0, float bv1,
                                          v8f& va, v8f& vb,
                                          _Float16* oBuf, int nt0, int nt1,
                                          int lane) {
  v8f c0 = {}, c1 = {};
#pragma unroll 4
  for (int kk = 0; kk < KST; ++kk) {
    v16h a = load_a_frag(aBuf, lane, kk);
    const v16h* w0 = (const v16h*)(wP + ((nt0 * KST + kk) * 32 + lane) * 16);
    const v16h* w1 = (const v16h*)(wP + ((nt1 * KST + kk) * 32 + lane) * 16);
    if (kk + 1 < KST) {  // prefetch next K-tile of the L2-hot weight stream
      __builtin_prefetch(wP + ((nt0 * KST + kk + 1) * 32 + lane) * 16, 0, 1);
      __builtin_prefetch(wP + ((nt1 * KST + kk + 1) * 32 + lane) * 16, 0, 1);
    }
    c0 = wmma_f16(a, *w0, c0);
    c1 = wmma_f16(a, *w1, c1);
  }
  lif_store(c0, bv0, va, oBuf, nt0, lane);
  lif_store(c1, bv1, vb, oBuf, nt1, lane);
}

// ---------------------------------------------------------------------------
// Main persistent kernel: one block = 16 batch rows for all 100 timesteps.
// 512 threads = 16 waves; wave w owns output columns [32w, 32w+32).
// ---------------------------------------------------------------------------
__global__ void __launch_bounds__(512, 1)
snn_forward(const float* __restrict__ x,
            const float* __restrict__ b1,
            const float* __restrict__ b2,
            const float* __restrict__ b3,
            const float* __restrict__ br,
            const float* __restrict__ tau_m,
            const _Float16* __restrict__ ws,
            float* __restrict__ out) {
  __shared__ __align__(16) _Float16 bufA[16 * LSTR];
  __shared__ __align__(16) _Float16 bufB[16 * LSTR];
  __shared__ float logitsS[16 * 16];

  const int tid  = threadIdx.x;
  const int wave = tid >> 5;
  const int lane = tid & 31;
  const int b0   = blockIdx.x * 16;

  const _Float16* W1p = ws;
  const _Float16* W2p = ws + KP1 * HDIM;
  const _Float16* W3p = W2p + HDIM * HDIM;
  const _Float16* Wrp = W3p + HDIM * HDIM;

  const int nt0  = wave * 2;
  const int nt1  = nt0 + 1;
  const int colL = lane & 15;

  // Per-lane biases for this wave's two column tiles (constant over t).
  const float b1v0 = b1[nt0 * 16 + colL], b1v1 = b1[nt1 * 16 + colL];
  const float b2v0 = b2[nt0 * 16 + colL], b2v1 = b2[nt1 * 16 + colL];
  const float b3v0 = b3[nt0 * 16 + colL], b3v1 = b3[nt1 * 16 + colL];
  // Readout constants (wave 0 tile only; cols >= DOUT padded harmlessly).
  const float brv   = (colL < DOUT) ? br[colL] : 0.0f;
  const float alpha = (colL < DOUT) ? 1.0f / (1.0f + __expf(-tau_m[colL])) : 0.0f;

  // Membrane states live in accumulator registers for the whole scan.
  v8f v1a = {}, v1b = {}, v2a = {}, v2b = {}, v3a = {}, v3b = {};
  v8f mem = {}, acc = {};

  for (int t = 0; t < TSTEPS; ++t) {
    // Opaque (always-0) per-iteration offset: forces fresh global_load_b128
    // of the packed weights each timestep instead of cross-t hoist + spill.
    const size_t z = opaque_zero();
    const _Float16* w1t = W1p + z;
    const _Float16* w2t = W2p + z;
    const _Float16* w3t = W3p + z;
    const _Float16* wrt = Wrp + z;

    // --- stage x_t tile [16 x 128] (f32 -> f16, zero-pad cols 120..127) ---
#pragma unroll
    for (int p = 0; p < 4; ++p) {
      int slot = tid + p * 512;
      int row  = slot >> 7;
      int col  = slot & 127;
      float xv = (col < M_IN)
                     ? x[(size_t)(b0 + row) * (TSTEPS * M_IN) + t * M_IN + col]
                     : 0.0f;
      bufA[row * LSTR + col] = (_Float16)xv;
    }
    __syncthreads();

    // --- layer 1: K=128, A=x tile (bufA) -> spikes s1 (bufB) ---
    layer_fwd<KP1 / 32>(bufA, w1t, b1v0, b1v1, v1a, v1b, bufB, nt0, nt1, lane);
    __syncthreads();

    // --- layer 2: K=512, A=s1 (bufB) -> s2 (bufA) ---
    layer_fwd<HDIM / 32>(bufB, w2t, b2v0, b2v1, v2a, v2b, bufA, nt0, nt1, lane);
    __syncthreads();

    // --- layer 3: K=512, A=s2 (bufA) -> s3 (bufB) ---
    layer_fwd<HDIM / 32>(bufA, w3t, b3v0, b3v1, v3a, v3b, bufB, nt0, nt1, lane);
    __syncthreads();

    // --- readout + leaky integrator (wave 0, single 16x16 tile) ---
    if (wave == 0) {
      v8f c = {};
#pragma unroll 4
      for (int kk = 0; kk < HDIM / 32; ++kk) {
        v16h a = load_a_frag(bufB, lane, kk);
        const v16h* w = (const v16h*)(wrt + (kk * 32 + lane) * 16);
        c = wmma_f16(a, *w, c);
      }
#pragma unroll
      for (int r = 0; r < 8; ++r) {
        float o = c[r] + brv;                    // s3 @ Wr + br
        mem[r]  = mem[r] * alpha + (1.0f - alpha) * o;
        acc[r] += mem[r];
      }
    }
    __syncthreads();  // protect bufB from next iteration's layer-1 writes
  }

  // --- logits = mean over T; spill to LDS for row-wise log_softmax ---
  if (wave == 0) {
#pragma unroll
    for (int r = 0; r < 8; ++r) {
      int row = r + ((lane >> 4) << 3);
      logitsS[row * 16 + colL] = acc[r] * (1.0f / (float)TSTEPS);
    }
  }
  __syncthreads();

  if (tid < 16) {
    float m = -1e30f;
    for (int c = 0; c < DOUT; ++c) m = fmaxf(m, logitsS[tid * 16 + c]);
    float s = 0.0f;
    for (int c = 0; c < DOUT; ++c) s += __expf(logitsS[tid * 16 + c] - m);
    float ls = __logf(s);
    for (int c = 0; c < DOUT; ++c)
      out[(size_t)(b0 + tid) * DOUT + c] = logitsS[tid * 16 + c] - m - ls;
  }
}

// ---------------------------------------------------------------------------
extern "C" void kernel_launch(void* const* d_in, const int* in_sizes, int n_in,
                              void* d_out, int out_size, void* d_ws, size_t ws_size,
                              hipStream_t stream) {
  const float* x    = (const float*)d_in[0];
  const float* W1   = (const float*)d_in[1];
  const float* b1   = (const float*)d_in[2];
  const float* W2   = (const float*)d_in[3];
  const float* b2   = (const float*)d_in[4];
  const float* W3   = (const float*)d_in[5];
  const float* b3   = (const float*)d_in[6];
  const float* Wr   = (const float*)d_in[7];
  const float* br   = (const float*)d_in[8];
  const float* tau  = (const float*)d_in[9];
  _Float16* ws      = (_Float16*)d_ws;
  float* out        = (float*)d_out;

  const int prepN = KP1 * HDIM + 2 * HDIM * HDIM + HDIM * 16;  // 598016
  prep_pack<<<(prepN + 255) / 256, 256, 0, stream>>>(W1, W2, W3, Wr, ws);

  snn_forward<<<B_TOT / 16, 512, 0, stream>>>(x, b1, b2, b3, br, tau, ws, out);
}
